// LSTMEncoderDecoder_38268158607752
// MI455X (gfx1250) — compile-verified, matrix operand order checked by reading, not run on hardware
//
#include <hip/hip_runtime.h>

// ---------------------------------------------------------------------------
// Types for CDNA5 WMMA (gfx1250, wave32)
// ---------------------------------------------------------------------------
typedef __attribute__((ext_vector_type(16))) __bf16 v16bf;
typedef __attribute__((ext_vector_type(8)))  float  v8f;

union FragU { v16bf v; uint4 q[2]; };

__device__ __forceinline__ v8f wmma_bf16(v16bf a, v16bf b, v8f c) {
  // D = A(16x32 bf16) * B(32x16 bf16) + C(16x16 f32)
  return __builtin_amdgcn_wmma_f32_16x16x32_bf16(
      /*neg_a=*/false, a, /*neg_b=*/false, b,
      /*c_mod=*/(short)0, c, /*reuse_a=*/false, /*reuse_b=*/false);
}

// Round-to-nearest-even f32 -> bf16 (stored as ushort)
__device__ __forceinline__ unsigned short f2bf(float f) {
  unsigned u = __float_as_uint(f);
  unsigned r = u + 0x7FFFu + ((u >> 16) & 1u);
  return (unsigned short)(r >> 16);
}

__device__ __forceinline__ float sigmoidf_(float x) {
  return 1.0f / (1.0f + __expf(-x));
}

// Load one 16-bit A/B fragment per ISA 7.12.2 layout:
//   lane L: K chunk = ks*32 + (L>>4)*8,
//   VGPRs 0..3 = K[k0..k0+7], VGPRs 4..7 = K[k0+16..k0+23]  (two b128 loads)
__device__ __forceinline__ v16bf frag_ld(const unsigned short* base, int row,
                                         int K, int ks) {
  const int lane = threadIdx.x & 31;
  const int k0 = ks * 32 + ((lane >> 4) << 3);
  const unsigned short* p = base + row * K + k0;
  FragU u;
  u.q[0] = *(const uint4*)(p);
  u.q[1] = *(const uint4*)(p + 16);
  return u.v;
}

// ---------------------------------------------------------------------------
// Shared LSTM step (256 threads = 8 waves). Wave w owns hidden columns
// [16w,16w+16) of all 4 gates => cell state c stays in registers/locals.
// z = e[32,64] @ Wih^T + h[32,128] @ Whh^T + bsum ; gates i,f,g,o.
// h_new (== h_lds, guarded by internal barrier) gets bf16 of new h.
// ---------------------------------------------------------------------------
__device__ __forceinline__ void lstm_step(const unsigned short* e_lds,
                                          const unsigned short* h_lds,
                                          const unsigned short* Wih,
                                          const unsigned short* Whh,
                                          const float* bsum,
                                          float c0[8], float c1[8],
                                          unsigned short* h_new) {
  const int tid = threadIdx.x;
  const int wid = tid >> 5, lane = tid & 31;
  const int col = lane & 15, half = lane >> 4;
  const int m0 = col, m1 = 16 + col;   // A-matrix (batch) rows for mt0/mt1
  const int hh = wid * 16 + col;       // hidden column owned by this lane

  v8f ai0 = {}, af0 = {}, ag0 = {}, ao0 = {};
  v8f ai1 = {}, af1 = {}, ag1 = {}, ao1 = {};

#pragma unroll
  for (int ks = 0; ks < 2; ++ks) {     // K = E = 64
    v16bf a0 = frag_ld(e_lds, m0, 64, ks);
    v16bf a1 = frag_ld(e_lds, m1, 64, ks);
    v16bf bi = frag_ld(Wih,       hh, 64, ks);
    v16bf bf = frag_ld(Wih, 128 + hh, 64, ks);
    v16bf bg = frag_ld(Wih, 256 + hh, 64, ks);
    v16bf bo = frag_ld(Wih, 384 + hh, 64, ks);
    ai0 = wmma_bf16(a0, bi, ai0);  ai1 = wmma_bf16(a1, bi, ai1);
    af0 = wmma_bf16(a0, bf, af0);  af1 = wmma_bf16(a1, bf, af1);
    ag0 = wmma_bf16(a0, bg, ag0);  ag1 = wmma_bf16(a1, bg, ag1);
    ao0 = wmma_bf16(a0, bo, ao0);  ao1 = wmma_bf16(a1, bo, ao1);
  }
#pragma unroll
  for (int ks = 0; ks < 4; ++ks) {     // K = H = 128
    v16bf a0 = frag_ld(h_lds, m0, 128, ks);
    v16bf a1 = frag_ld(h_lds, m1, 128, ks);
    v16bf bi = frag_ld(Whh,       hh, 128, ks);
    v16bf bf = frag_ld(Whh, 128 + hh, 128, ks);
    v16bf bg = frag_ld(Whh, 256 + hh, 128, ks);
    v16bf bo = frag_ld(Whh, 384 + hh, 128, ks);
    ai0 = wmma_bf16(a0, bi, ai0);  ai1 = wmma_bf16(a1, bi, ai1);
    af0 = wmma_bf16(a0, bf, af0);  af1 = wmma_bf16(a1, bf, af1);
    ag0 = wmma_bf16(a0, bg, ag0);  ag1 = wmma_bf16(a1, bg, ag1);
    ao0 = wmma_bf16(a0, bo, ao0);  ao1 = wmma_bf16(a1, bo, ao1);
  }

  // All waves done reading e_lds/h_lds before anyone overwrites h.
  __syncthreads();

  const float bi_s = bsum[hh], bf_s = bsum[128 + hh];
  const float bg_s = bsum[256 + hh], bo_s = bsum[384 + hh];
#pragma unroll
  for (int v = 0; v < 8; ++v) {
    const int b0 = v + 8 * half, b1 = 16 + b0;  // C/D layout: vgpr v, lane half
    float ig = sigmoidf_(ai0[v] + bi_s);
    float fg = sigmoidf_(af0[v] + bf_s);
    float gg = tanhf(ag0[v] + bg_s);
    float og = sigmoidf_(ao0[v] + bo_s);
    float c = fg * c0[v] + ig * gg;
    c0[v] = c;
    h_new[b0 * 128 + hh] = f2bf(og * tanhf(c));

    ig = sigmoidf_(ai1[v] + bi_s);
    fg = sigmoidf_(af1[v] + bf_s);
    gg = tanhf(ag1[v] + bg_s);
    og = sigmoidf_(ao1[v] + bo_s);
    c = fg * c1[v] + ig * gg;
    c1[v] = c;
    h_new[b1 * 128 + hh] = f2bf(og * tanhf(c));
  }
}

// ---------------------------------------------------------------------------
// Kernel 1: weight conversion f32 -> bf16, fused bias sums, zero key slots
// ---------------------------------------------------------------------------
__global__ void __launch_bounds__(256)
convert_kernel(const float* eWih, const float* eWhh, const float* dWih,
               const float* dWhh, const float* fcW, const float* ebih,
               const float* ebhh, const float* dbih, const float* dbhh,
               unsigned short* o_eWih, unsigned short* o_eWhh,
               unsigned short* o_dWih, unsigned short* o_dWhh,
               unsigned short* o_fcW, float* o_eb, float* o_db,
               unsigned long long* keys_arr) {
  const int idx = blockIdx.x * blockDim.x + threadIdx.x;
  const int stride = gridDim.x * blockDim.x;
  for (int i = idx; i < 32000 * 128; i += stride) o_fcW[i] = f2bf(fcW[i]);
  for (int i = idx; i < 512 * 64; i += stride) {
    o_eWih[i] = f2bf(eWih[i]);
    o_dWih[i] = f2bf(dWih[i]);
  }
  for (int i = idx; i < 512 * 128; i += stride) {
    o_eWhh[i] = f2bf(eWhh[i]);
    o_dWhh[i] = f2bf(dWhh[i]);
  }
  for (int i = idx; i < 512; i += stride) {
    o_eb[i] = ebih[i] + ebhh[i];
    o_db[i] = dbih[i] + dbhh[i];
  }
  for (int i = idx; i < 128 * 32; i += stride) keys_arr[i] = 0ull;
}

// ---------------------------------------------------------------------------
// Kernel 2: encoder — one workgroup, 128 sequential steps, c in registers
// ---------------------------------------------------------------------------
__global__ void __launch_bounds__(256)
enc_kernel(const int* x, const float* embed, const unsigned short* Wih,
           const unsigned short* Whh, const float* bsum, float* c_out,
           unsigned short* h_out) {
  __shared__ __align__(16) unsigned short e_lds[32 * 64];
  __shared__ __align__(16) unsigned short h_lds[32 * 128];
  const int tid = threadIdx.x;

  float c0[8], c1[8];
#pragma unroll
  for (int v = 0; v < 8; ++v) { c0[v] = 0.0f; c1[v] = 0.0f; }
  for (int i = tid; i < 32 * 128; i += 256) h_lds[i] = 0;
  __syncthreads();

  for (int t = 0; t < 128; ++t) {
    // Embedding lookup (padding_idx = 0 -> zero row)
    for (int i = tid; i < 32 * 64; i += 256) {
      const int b = i >> 6, k = i & 63;
      const int tok = x[b * 128 + t];
      e_lds[i] = (tok == 0) ? (unsigned short)0 : f2bf(embed[tok * 64 + k]);
    }
    __syncthreads();
    lstm_step(e_lds, h_lds, Wih, Whh, bsum, c0, c1, h_lds);
    __syncthreads();
  }

  // Export final (h, c) for the decoder (parity buffer 0)
  const int wid = tid >> 5, lane = tid & 31;
  const int col = lane & 15, half = lane >> 4;
  const int hh = wid * 16 + col;
#pragma unroll
  for (int v = 0; v < 8; ++v) {
    c_out[(v + 8 * half) * 128 + hh] = c0[v];
    c_out[(16 + v + 8 * half) * 128 + hh] = c1[v];
  }
  for (int i = tid; i < 32 * 128; i += 256) h_out[i] = h_lds[i];
}

// ---------------------------------------------------------------------------
// Argmax key helpers: key = mono(logit)<<32 | ~n  (u64 max => argmax, ties
// resolve to the lowest vocab index, matching jnp.argmax)
// ---------------------------------------------------------------------------
__device__ __forceinline__ unsigned long long pack_key(float f, int n) {
  unsigned u = __float_as_uint(f);
  u = (u & 0x80000000u) ? ~u : (u | 0x80000000u);  // monotonic float map
  return ((unsigned long long)u << 32) | (unsigned)(~n);
}

__device__ __forceinline__ unsigned long long shflxor16_u64(
    unsigned long long v, int d) {
  unsigned lo = (unsigned)v, hi = (unsigned)(v >> 32);
  lo = __shfl_xor(lo, d, 16);
  hi = __shfl_xor(hi, d, 16);
  return ((unsigned long long)hi << 32) | lo;
}

// ---------------------------------------------------------------------------
// Kernel 3: fused decoder step. Each of the 250 blocks redundantly computes
// the tiny LSTM cell from read-only (h_in, c_in, keys_prev); block 0 alone
// publishes (h,c) into the opposite parity buffer. Then each wave computes
// one 16-column tile of logits = h @ fc_W^T + b (A-frags straight from LDS),
// writes d_out, and reduces argmax into this step's dedicated key slot.
// grid = 250 x 256 threads (8 waves); wave -> N-tile blockIdx*8+wid.
// ---------------------------------------------------------------------------
__global__ void __launch_bounds__(256)
step_kernel(int t, const float* embed, const unsigned short* Wih,
            const unsigned short* Whh, const float* bsum, const float* c_in,
            float* c_out, const unsigned short* h_in, unsigned short* h_out,
            const unsigned short* fcW, const float* fcb, float* out,
            const unsigned long long* keys_prev, unsigned long long* keys_cur) {
  __shared__ __align__(16) unsigned short e_lds[32 * 64];
  __shared__ __align__(16) unsigned short h_lds[32 * 128];
  __shared__ int stok[32];
  __shared__ unsigned long long lkeys[32];
  const int tid = threadIdx.x;

  if (tid < 32) {
    stok[tid] = (t == 0) ? 1 : (int)(~(unsigned)keys_prev[tid]);  // SOS = 1
    lkeys[tid] = 0ull;
  }
  for (int i = tid; i < 32 * 128; i += 256) h_lds[i] = h_in[i];
  __syncthreads();

  // Embedding lookup for the fed-back token (padding_idx = 0 -> zero row)
  for (int i = tid; i < 32 * 64; i += 256) {
    const int b = i >> 6, k = i & 63;
    const int tok = stok[b];
    e_lds[i] = (tok == 0) ? (unsigned short)0 : f2bf(embed[tok * 64 + k]);
  }

  const int wid = tid >> 5, lane = tid & 31;
  const int col = lane & 15, half = lane >> 4;
  const int hh = wid * 16 + col;
  float c0[8], c1[8];
#pragma unroll
  for (int v = 0; v < 8; ++v) {
    c0[v] = c_in[(v + 8 * half) * 128 + hh];
    c1[v] = c_in[(16 + v + 8 * half) * 128 + hh];
  }
  __syncthreads();

  lstm_step(e_lds, h_lds, Wih, Whh, bsum, c0, c1, h_lds);
  __syncthreads();

  // Only block 0 publishes state for step t+1 (opposite parity buffers).
  if (blockIdx.x == 0) {
#pragma unroll
    for (int v = 0; v < 8; ++v) {
      c_out[(v + 8 * half) * 128 + hh] = c0[v];
      c_out[(16 + v + 8 * half) * 128 + hh] = c1[v];
    }
    for (int i = tid; i < 32 * 128; i += 256) h_out[i] = h_lds[i];
  }

  // ----- logits tile: N-tile per wave, h fragments from LDS -----
  const int nt = blockIdx.x * 8 + wid;  // 0..1999
  const int n = nt * 16 + col;          // vocab column for this lane

  v8f acc0 = {}, acc1 = {};
#pragma unroll
  for (int ks = 0; ks < 4; ++ks) {  // K = 128
    v16bf a0 = frag_ld(h_lds, col, 128, ks);       // batches 0..15
    v16bf a1 = frag_ld(h_lds, 16 + col, 128, ks);  // batches 16..31
    v16bf b = frag_ld(fcW, n, 128, ks);
    acc0 = wmma_bf16(a0, b, acc0);
    acc1 = wmma_bf16(a1, b, acc1);
  }

  const float bias = fcb[n];
  unsigned long long k0[8], k1[8];
#pragma unroll
  for (int v = 0; v < 8; ++v) {
    const int b0 = v + 8 * half, b1 = 16 + b0;
    const float f0 = acc0[v] + bias;
    const float f1 = acc1[v] + bias;
    out[((size_t)b0 * 128 + t) * 32000 + n] = f0;
    out[((size_t)b1 * 128 + t) * 32000 + n] = f1;
    k0[v] = pack_key(f0, n);
    k1[v] = pack_key(f1, n);
  }

  // Butterfly max over the 16 lanes of each half (same batch rows per half)
#pragma unroll
  for (int d = 1; d < 16; d <<= 1) {
#pragma unroll
    for (int v = 0; v < 8; ++v) {
      unsigned long long o0 = shflxor16_u64(k0[v], d);
      unsigned long long o1 = shflxor16_u64(k1[v], d);
      if (o0 > k0[v]) k0[v] = o0;
      if (o1 > k1[v]) k1[v] = o1;
    }
  }
  if ((lane & 15) == 0) {  // lanes 0 and 16 hold the reduced keys
#pragma unroll
    for (int v = 0; v < 8; ++v) {
      atomicMax(&lkeys[v + 8 * half], k0[v]);
      atomicMax(&lkeys[16 + v + 8 * half], k1[v]);
    }
  }
  __syncthreads();
  if (tid < 32) atomicMax(&keys_cur[tid], lkeys[tid]);
}

// ---------------------------------------------------------------------------
// Host launch: convert + encoder + 128 fused decoder steps = 130 graph nodes
// ---------------------------------------------------------------------------
extern "C" void kernel_launch(void* const* d_in, const int* in_sizes, int n_in,
                              void* d_out, int out_size, void* d_ws,
                              size_t ws_size, hipStream_t stream) {
  (void)in_sizes; (void)n_in; (void)out_size; (void)ws_size;
  const int*   x     = (const int*)d_in[0];
  const float* embed = (const float*)d_in[1];
  const float* eWih  = (const float*)d_in[2];
  const float* eWhh  = (const float*)d_in[3];
  const float* ebih  = (const float*)d_in[4];
  const float* ebhh  = (const float*)d_in[5];
  const float* dWih  = (const float*)d_in[6];
  const float* dWhh  = (const float*)d_in[7];
  const float* dbih  = (const float*)d_in[8];
  const float* dbhh  = (const float*)d_in[9];
  const float* fcW   = (const float*)d_in[10];
  const float* fcb   = (const float*)d_in[11];
  float* out = (float*)d_out;

  char* ws = (char*)d_ws;
  unsigned short* fcW_bf   = (unsigned short*)(ws);            // 8,192,000 B
  unsigned short* eWih_bf  = (unsigned short*)(ws + 8192000);  // 65,536 B
  unsigned short* eWhh_bf  = (unsigned short*)(ws + 8257536);  // 131,072 B
  unsigned short* dWih_bf  = (unsigned short*)(ws + 8388608);  // 65,536 B
  unsigned short* dWhh_bf  = (unsigned short*)(ws + 8454144);  // 131,072 B
  float* eb_sum            = (float*)(ws + 8585216);           // 2,048 B
  float* db_sum            = (float*)(ws + 8587264);           // 2,048 B
  float* c_buf0            = (float*)(ws + 8589312);           // 16,384 B
  float* c_buf1            = (float*)(ws + 8605696);           // 16,384 B
  unsigned short* h_buf0   = (unsigned short*)(ws + 8622080);  // 8,192 B
  unsigned short* h_buf1   = (unsigned short*)(ws + 8630272);  // 8,192 B
  unsigned long long* keys = (unsigned long long*)(ws + 8638464);  // 32,768 B

  convert_kernel<<<2048, 256, 0, stream>>>(
      eWih, eWhh, dWih, dWhh, fcW, ebih, ebhh, dbih, dbhh,
      eWih_bf, eWhh_bf, dWih_bf, dWhh_bf, fcW_bf, eb_sum, db_sum, keys);

  enc_kernel<<<1, 256, 0, stream>>>(x, embed, eWih_bf, eWhh_bf, eb_sum,
                                    c_buf0, h_buf0);

  for (int t = 0; t < 128; ++t) {
    const float* c_in = (t & 1) ? c_buf1 : c_buf0;
    float* c_out = (t & 1) ? c_buf0 : c_buf1;
    const unsigned short* h_in = (t & 1) ? h_buf1 : h_buf0;
    unsigned short* h_out = (t & 1) ? h_buf0 : h_buf1;
    const unsigned long long* keys_prev = keys + 32 * (t > 0 ? t - 1 : 0);
    unsigned long long* keys_cur = keys + 32 * t;
    step_kernel<<<250, 256, 0, stream>>>(
        t, embed, dWih_bf, dWhh_bf, db_sum, c_in, c_out, h_in, h_out,
        fcW_bf, fcb, out, keys_prev, keys_cur);
  }
}
